// DiffusionAttentionBlock_12008728560322
// MI455X (gfx1250) — matrix operand order repaired and checked by reading, not compiled, not run on hardware
//
#include <hip/hip_runtime.h>
#include <hip/hip_bf16.h>

// ---------------------------------------------------------------------------
// Types / WMMA helpers (CDNA5 gfx1250, wave32)
// ---------------------------------------------------------------------------
typedef __attribute__((ext_vector_type(16))) _Float16 v16h;
typedef __attribute__((ext_vector_type(8)))  _Float16 v8h;
typedef __attribute__((ext_vector_type(8)))  float    v8f;
typedef __attribute__((ext_vector_type(4)))  unsigned int u32x4;
typedef __attribute__((ext_vector_type(8)))  int          i32x8;
typedef __attribute__((ext_vector_type(4)))  int          i32x4;

#define DEV __device__ __forceinline__

DEV v8f wmma16(v16h a, v16h b, v8f c) {
  // D = A(16x32 f16) * B(32x16 f16) + C(16x16 f32)
  return __builtin_amdgcn_wmma_f32_16x16x32_f16(false, a, false, b, (short)0, c,
                                                false, false);
}

// A fragment: 16x32 f16 from row-major [16 x >=32] buffer, row stride ld (mult of 8).
// Lane m=lane&15, khalf=(lane>>4)*8.  Elements 0..7 = K kh+0..7, 8..15 = K kh+16..23.
// Two 16-byte LDS loads per lane (ds_load_b128).
DEV v16h load_a_frag(const _Float16* base, int ld) {
  int lane = threadIdx.x & 31;
  const _Float16* p = base + (lane & 15) * ld + ((lane >> 4) << 3);
  v8h lo = *(const v8h*)p;
  v8h hi = *(const v8h*)(p + 16);
  v16h a;
#pragma unroll
  for (int i = 0; i < 8; ++i) { a[i] = lo[i]; a[8 + i] = hi[i]; }
  return a;
}

// B fragment computing A @ src^T, where src is row-major [16 x >=32] with the
// WMMA "N" dimension as rows and K contiguous: lane n=lane&15, kh=(lane>>4)*8,
// elements 0..7 = K kh+0..7, 8..15 = K kh+16..23.  Two 16-byte LDS loads per lane.
DEV v16h load_bt_frag(const _Float16* base, int ld) {
  int lane = threadIdx.x & 31;
  const _Float16* p = base + (lane & 15) * ld + ((lane >> 4) << 3);
  v8h lo = *(const v8h*)p;
  v8h hi = *(const v8h*)(p + 16);
  v16h b;
#pragma unroll
  for (int i = 0; i < 8; ++i) { b[i] = lo[i]; b[8 + i] = hi[i]; }
  return b;
}

DEV float gelu_exact(float v) { return 0.5f * v * (1.0f + erff(v * 0.70710678118f)); }

// ---------------------------------------------------------------------------
// Weight conversion fp32 [K][N] -> f16 transposed [N][K]
// ---------------------------------------------------------------------------
__global__ void k_cvt_f16_t(const float* __restrict__ src, _Float16* __restrict__ dst,
                            int K, int N) {
  int i = blockIdx.x * 256 + threadIdx.x;
  if (i < K * N) {
    int k = i / N, n = i % N;
    dst[(size_t)n * K + k] = (_Float16)src[i];
  }
}

// ---------------------------------------------------------------------------
// Precompute G = K4 @ viewmat and Ginv per (b,cam): 16 matrices of 4x4.
// ---------------------------------------------------------------------------
__global__ void k_gmats(const float* __restrict__ vm, const float* __restrict__ Ks,
                        float* __restrict__ Gm, float* __restrict__ Gi) {
  int t = threadIdx.x;
  if (t >= 16) return;
  const float* V = vm + t * 16;
  const float* Kp = Ks + t * 9;
  float G[16];
#pragma unroll
  for (int i = 0; i < 3; ++i)
#pragma unroll
    for (int j = 0; j < 4; ++j)
      G[i * 4 + j] = Kp[i * 3 + 0] * V[0 * 4 + j] + Kp[i * 3 + 1] * V[1 * 4 + j] +
                     Kp[i * 3 + 2] * V[2 * 4 + j];
  G[12] = 0.f; G[13] = 0.f; G[14] = 0.f; G[15] = 1.f;
  float a = G[0], b = G[1], c = G[2];
  float d = G[4], e = G[5], f = G[6];
  float g = G[8], h = G[9], i2 = G[10];
  float det = a * (e * i2 - f * h) - b * (d * i2 - f * g) + c * (d * h - e * g);
  float rd = 1.0f / det;
  float Ai[9];
  Ai[0] =  (e * i2 - f * h) * rd; Ai[1] = -(b * i2 - c * h) * rd; Ai[2] =  (b * f - c * e) * rd;
  Ai[3] = -(d * i2 - f * g) * rd; Ai[4] =  (a * i2 - c * g) * rd; Ai[5] = -(a * f - c * d) * rd;
  Ai[6] =  (d * h - e * g) * rd;  Ai[7] = -(a * h - b * g) * rd;  Ai[8] =  (a * e - b * d) * rd;
  float bx = G[3], by = G[7], bz = G[11];
  float GI[16];
#pragma unroll
  for (int i = 0; i < 3; ++i) {
    GI[i * 4 + 0] = Ai[i * 3 + 0];
    GI[i * 4 + 1] = Ai[i * 3 + 1];
    GI[i * 4 + 2] = Ai[i * 3 + 2];
    GI[i * 4 + 3] = -(Ai[i * 3 + 0] * bx + Ai[i * 3 + 1] * by + Ai[i * 3 + 2] * bz);
  }
  GI[12] = 0.f; GI[13] = 0.f; GI[14] = 0.f; GI[15] = 1.f;
#pragma unroll
  for (int i = 0; i < 16; ++i) { Gm[t * 16 + i] = G[i]; Gi[t * 16 + i] = GI[i]; }
}

// ---------------------------------------------------------------------------
// LayerNorm (768): one wave per row, 8 rows per block, f16 output.
// ---------------------------------------------------------------------------
__global__ __launch_bounds__(256) void k_ln(const float* __restrict__ x,
                                            const float* __restrict__ g,
                                            const float* __restrict__ b,
                                            _Float16* __restrict__ out) {
  int w = threadIdx.x >> 5, lane = threadIdx.x & 31;
  int row = blockIdx.x * 8 + w;
  const float* xr = x + (size_t)row * 768;
  float vals[24];
  float s = 0.f;
#pragma unroll
  for (int i = 0; i < 24; ++i) { float v = xr[lane + i * 32]; vals[i] = v; s += v; }
#pragma unroll
  for (int off = 1; off < 32; off <<= 1) s += __shfl_xor(s, off, 32);
  float mean = s * (1.0f / 768.0f);
  float s2 = 0.f;
#pragma unroll
  for (int i = 0; i < 24; ++i) { float d = vals[i] - mean; s2 += d * d; }
#pragma unroll
  for (int off = 1; off < 32; off <<= 1) s2 += __shfl_xor(s2, off, 32);
  float rs = rsqrtf(s2 * (1.0f / 768.0f) + 1e-5f);
#pragma unroll
  for (int i = 0; i < 24; ++i) {
    int c = lane + i * 32;
    out[(size_t)row * 768 + c] = (_Float16)((vals[i] - mean) * rs * g[c] + b[c]);
  }
}

// ---------------------------------------------------------------------------
// WMMA GEMM: C[M,N] = epilogue(A16[M,K] @ Bt16[N,K]^T).
// 256 threads = 8 waves; block tile 64x128; wave = 16 rows x 64 cols
// (1 A frag + 4 B frags + 4 WMMA per 32-wide K step).
// flags: 1=GELU, 2=f16 out, 4=+residual(f32), 8=+bias.
// Also issues one TDM tensor_load_to_lds per block (guarded; L2-warming
// prefetch of this block's weight panel into scratch LDS).
// ---------------------------------------------------------------------------
#define LDT_S 40
__global__ __launch_bounds__(256) void k_gemm(const _Float16* __restrict__ A,
                                              const _Float16* __restrict__ Bt,
                                              const float* __restrict__ bias,
                                              const float* __restrict__ res,
                                              void* __restrict__ Cout,
                                              int M, int N, int K, int flags) {
  __shared__ _Float16 As[64 * LDT_S];
  __shared__ _Float16 BsT[128 * LDT_S];
#if __has_builtin(__builtin_amdgcn_tensor_load_to_lds)
  __shared__ _Float16 TdmScratch[2048];
#endif
  int tid = threadIdx.x;
  int bm = blockIdx.y * 64;
  int bn = blockIdx.x * 128;
  int w = tid >> 5, wr = w & 3, wc = w >> 2;
  int arow = tid >> 2, acol = (tid & 3) * 8;
  int brow = tid >> 1, bk = (tid & 1) * 16;
  v8f acc[4] = {v8f{}, v8f{}, v8f{}, v8f{}};

#if __has_builtin(__builtin_amdgcn_tensor_load_to_lds) && \
    __has_builtin(__builtin_amdgcn_s_wait_tensorcnt)
  if (tid == 0) {
    unsigned lds_off =
        (unsigned)(unsigned long long)(__attribute__((address_space(3))) _Float16*)TdmScratch;
    unsigned long long ga = (unsigned long long)(const void*)(Bt + (size_t)bn * K);
    u32x4 g0;
    g0[0] = 1u;                                   // count=1 valid descriptor
    g0[1] = lds_off;                              // lds_addr
    g0[2] = (unsigned)(ga & 0xffffffffu);         // global_addr[31:0]
    g0[3] = (unsigned)((ga >> 32) & 0x01ffffffu)  // global_addr[56:32]
            | (2u << 30);                         // type=2 (image)
    i32x8 g1;
    g1[0] = (1 << 16);                            // data_size = 2 bytes
    g1[1] = (int)(64u << 16);                     // tensor_dim0 = 64 (low16)
    g1[2] = (int)(32u << 16);                     // tensor_dim1 = 32 (low16)
    g1[3] = (int)(64u << 16);                     // tile_dim0 = 64
    g1[4] = 32;                                   // tile_dim1 = 32, tile_dim2 = 0
    g1[5] = (int)(unsigned)K;                     // tensor_dim0_stride low32
    g1[6] = 0;
    g1[7] = 0;
    i32x4 g2 = {0, 0, 0, 0};
    i32x4 g3 = {0, 0, 0, 0};
    i32x8 g4 = {0, 0, 0, 0, 0, 0, 0, 0};
    __builtin_amdgcn_tensor_load_to_lds(g0, g1, g2, g3, g4, 0);
  }
  __builtin_amdgcn_s_wait_tensorcnt((short)0);
#endif

  for (int k0 = 0; k0 < K; k0 += 32) {
    *(v8h*)(As + arow * LDT_S + acol) =
        *(const v8h*)(A + (size_t)(bm + arow) * K + k0 + acol);
    *(v8h*)(BsT + brow * LDT_S + bk) =
        *(const v8h*)(Bt + (size_t)(bn + brow) * K + k0 + bk);
    *(v8h*)(BsT + brow * LDT_S + bk + 8) =
        *(const v8h*)(Bt + (size_t)(bn + brow) * K + k0 + bk + 8);
    if (k0 + 32 < K) {  // global_prefetch_b8 of next K panel
      __builtin_prefetch(A + (size_t)(bm + arow) * K + k0 + 32 + acol, 0, 1);
      __builtin_prefetch(Bt + (size_t)(bn + brow) * K + k0 + 32 + bk, 0, 1);
    }
    __syncthreads();
    v16h a = load_a_frag(As + wr * 16 * LDT_S, LDT_S);
#pragma unroll
    for (int t = 0; t < 4; ++t) {
      v16h b = load_bt_frag(BsT + (wc * 64 + t * 16) * LDT_S, LDT_S);
      acc[t] = wmma16(a, b, acc[t]);
    }
    __syncthreads();
  }
  int lane = tid & 31, nn = lane & 15, mh = (lane >> 4) * 8;
#pragma unroll
  for (int t = 0; t < 4; ++t) {
    int col = bn + wc * 64 + t * 16 + nn;
    float bv = (flags & 8) ? bias[col] : 0.0f;
#pragma unroll
    for (int r = 0; r < 8; ++r) {
      int rowg = bm + wr * 16 + mh + r;
      float v = acc[t][r] + bv;
      if (flags & 1) v = gelu_exact(v);
      if (flags & 4) v += res[(size_t)rowg * N + col];
      if (flags & 2) ((_Float16*)Cout)[(size_t)rowg * N + col] = (_Float16)v;
      else           ((float*)Cout)[(size_t)rowg * N + col] = v;
    }
  }
}

// ---------------------------------------------------------------------------
// Local per-view attention with interleaved RoPE.  One block per (view,head):
// 192 blocks, 256 threads (8 waves).  P=256 tokens, DH=64.
// LDS: qs/ks [256x64 f16] (K contiguous), vsT [64 x 264 f16] (transposed so
// the P@V B-fragments are contiguous), per-wave P staging [16x32 f16].
// ---------------------------------------------------------------------------
#define LDV_L 264
extern __shared__ __attribute__((aligned(16))) char smem[];

__global__ __launch_bounds__(256) void k_attn_local(const float* __restrict__ qkv,
                                                    _Float16* __restrict__ y) {
  _Float16* qs  = (_Float16*)smem;          // 256*64
  _Float16* ks  = qs + 256 * 64;            // 256*64
  _Float16* vsT = ks + 256 * 64;            // 64*LDV_L
  _Float16* pb  = vsT + 64 * LDV_L;         // 8 waves * 16*32
  int bc = blockIdx.x / 12, h = blockIdx.x % 12;
  int base = bc * 256;
  int tid = threadIdx.x;
  for (int idx = tid; idx < 256 * 32; idx += 256) {
    int row = idx >> 5, p = idx & 31;
    const float* src = qkv + (size_t)(base + row) * 2304 + h * 64 + p * 2;
    float inv = __expf(-((float)(2 * p) / 64.0f) * 9.210340371976184f);  // 10000^-x
    float cn, sn;
    __sincosf((float)row * inv, &sn, &cn);
    float q1 = src[0], q2 = src[1];
    qs[row * 64 + 2 * p]     = (_Float16)(q1 * cn - q2 * sn);
    qs[row * 64 + 2 * p + 1] = (_Float16)(q1 * sn + q2 * cn);
    float k1 = src[768], k2 = src[769];
    ks[row * 64 + 2 * p]     = (_Float16)(k1 * cn - k2 * sn);
    ks[row * 64 + 2 * p + 1] = (_Float16)(k1 * sn + k2 * cn);
    vsT[(2 * p) * LDV_L + row]     = (_Float16)src[1536];
    vsT[(2 * p + 1) * LDV_L + row] = (_Float16)src[1537];
  }
  __syncthreads();
  int w = tid >> 5, lane = tid & 31, nn = lane & 15, mh = (lane >> 4) * 8;
  _Float16* pbuf = pb + w * (16 * 32);
  for (int qt = w; qt < 16; qt += 8) {
    const _Float16* qb = qs + qt * 16 * 64;
    v8f sv[16];
#pragma unroll
    for (int kt = 0; kt < 16; ++kt) {
      v8f sa = {};
#pragma unroll
      for (int c = 0; c < 2; ++c) {
        v16h a = load_a_frag(qb + c * 32, 64);
        v16h b = load_bt_frag(ks + kt * 16 * 64 + c * 32, 64);
        sa = wmma16(a, b, sa);
      }
#pragma unroll
      for (int r = 0; r < 8; ++r) sa[r] *= 0.125f;
      sv[kt] = sa;
    }
    float lrow[8];
#pragma unroll
    for (int r = 0; r < 8; ++r) {
      float mx = -1e30f;
#pragma unroll
      for (int kt = 0; kt < 16; ++kt) mx = fmaxf(mx, sv[kt][r]);
#pragma unroll
      for (int off = 1; off < 16; off <<= 1) mx = fmaxf(mx, __shfl_xor(mx, off, 32));
      float sum = 0.f;
#pragma unroll
      for (int kt = 0; kt < 16; ++kt) {
        float e = __expf(sv[kt][r] - mx);
        sv[kt][r] = e;
        sum += e;
      }
#pragma unroll
      for (int off = 1; off < 16; off <<= 1) sum += __shfl_xor(sum, off, 32);
      lrow[r] = sum;
    }
    v8f ov[4] = {v8f{}, v8f{}, v8f{}, v8f{}};
#pragma unroll
    for (int kc = 0; kc < 8; ++kc) {
#pragma unroll
      for (int t = 0; t < 2; ++t) {
        v8f pv = sv[2 * kc + t];
#pragma unroll
        for (int r = 0; r < 8; ++r)
          pbuf[(mh + r) * 32 + t * 16 + nn] = (_Float16)pv[r];
      }
      v16h a = load_a_frag(pbuf, 32);
#pragma unroll
      for (int nt = 0; nt < 4; ++nt) {
        v16h b = load_bt_frag(vsT + (nt * 16) * LDV_L + kc * 32, LDV_L);
        ov[nt] = wmma16(a, b, ov[nt]);
      }
    }
#pragma unroll
    for (int nt = 0; nt < 4; ++nt)
#pragma unroll
      for (int r = 0; r < 8; ++r) {
        int row = base + qt * 16 + mh + r;
        y[(size_t)row * 768 + h * 64 + nt * 16 + nn] = (_Float16)(ov[nt][r] / lrow[r]);
      }
  }
}

// ---------------------------------------------------------------------------
// Global PRoPE attention, flash-style streaming over 2048 keys.
// One block per (b, head, 128-query chunk): 384 blocks, 8 waves.
// q -> Ginv^T q, k/v -> G k, out -> Ginv o.  V staged transposed (vsbT).
// ---------------------------------------------------------------------------
#define LDV_G 72
__global__ __launch_bounds__(256) void k_attn_global(const float* __restrict__ qkv,
                                                     const float* __restrict__ Gm,
                                                     const float* __restrict__ Gi,
                                                     _Float16* __restrict__ y) {
  _Float16* qs   = (_Float16*)smem;            // 128*64
  _Float16* ksb  = qs + 128 * 64;              // 64*64
  _Float16* vsbT = ksb + 64 * 64;              // 64*LDV_G
  _Float16* pb   = vsbT + 64 * LDV_G;          // 8*16*64
  float*    ob   = (float*)(pb + 8 * 16 * 64); // 8*16*64 f32
  int bi = blockIdx.x;
  int b = bi / 192;
  int h = (bi / 16) % 12;
  int qb0 = (bi & 15) * 128;
  int tid = threadIdx.x;
  for (int idx = tid; idx < 128 * 16; idx += 256) {
    int row = idx >> 4, g = idx & 15;
    int s = qb0 + row;
    const float* M = Gi + (size_t)(b * 8 + (s >> 8)) * 16;
    const float* src = qkv + (size_t)(b * 2048 + s) * 2304 + h * 64 + g * 4;
    float x0 = src[0], x1 = src[1], x2 = src[2], x3 = src[3];
#pragma unroll
    for (int i = 0; i < 4; ++i)
      qs[row * 64 + g * 4 + i] =
          (_Float16)(M[0 * 4 + i] * x0 + M[1 * 4 + i] * x1 + M[2 * 4 + i] * x2 + M[3 * 4 + i] * x3);
  }
  int w = tid >> 5, lane = tid & 31, nn = lane & 15, mh = (lane >> 4) * 8;
  _Float16* pbuf = pb + w * 16 * 64;
  float* obuf = ob + w * 16 * 64;
  float mrow[8], lrow[8];
  v8f ov[4] = {v8f{}, v8f{}, v8f{}, v8f{}};
#pragma unroll
  for (int r = 0; r < 8; ++r) { mrow[r] = -1e30f; lrow[r] = 0.f; }
  __syncthreads();
  const _Float16* qrow = qs + w * 16 * 64;
  for (int kb = 0; kb < 2048; kb += 64) {
    for (int idx = tid; idx < 64 * 16; idx += 256) {
      int row = idx >> 4, g = idx & 15;
      int s = kb + row;
      const float* M = Gm + (size_t)(b * 8 + (s >> 8)) * 16;
      const float* sk = qkv + (size_t)(b * 2048 + s) * 2304 + 768 + h * 64 + g * 4;
      const float* sv2 = sk + 768;
      float k0 = sk[0], k1 = sk[1], k2 = sk[2], k3 = sk[3];
      float v0 = sv2[0], v1 = sv2[1], v2 = sv2[2], v3 = sv2[3];
#pragma unroll
      for (int i = 0; i < 4; ++i) {
        ksb[row * 64 + g * 4 + i] =
            (_Float16)(M[i * 4 + 0] * k0 + M[i * 4 + 1] * k1 + M[i * 4 + 2] * k2 + M[i * 4 + 3] * k3);
        vsbT[(g * 4 + i) * LDV_G + row] =
            (_Float16)(M[i * 4 + 0] * v0 + M[i * 4 + 1] * v1 + M[i * 4 + 2] * v2 + M[i * 4 + 3] * v3);
      }
    }
    __syncthreads();
    v8f sb[4];
#pragma unroll
    for (int kt = 0; kt < 4; ++kt) {
      v8f sa = {};
#pragma unroll
      for (int c = 0; c < 2; ++c) {
        v16h a = load_a_frag(qrow + c * 32, 64);
        v16h bb = load_bt_frag(ksb + kt * 16 * 64 + c * 32, 64);
        sa = wmma16(a, bb, sa);
      }
#pragma unroll
      for (int r = 0; r < 8; ++r) sa[r] *= 0.125f;
      sb[kt] = sa;
    }
#pragma unroll
    for (int r = 0; r < 8; ++r) {
      float bm = -1e30f;
#pragma unroll
      for (int kt = 0; kt < 4; ++kt) bm = fmaxf(bm, sb[kt][r]);
#pragma unroll
      for (int off = 1; off < 16; off <<= 1) bm = fmaxf(bm, __shfl_xor(bm, off, 32));
      float mn = fmaxf(mrow[r], bm);
      float sc = __expf(mrow[r] - mn);
      mrow[r] = mn;
      float sum = 0.f;
#pragma unroll
      for (int kt = 0; kt < 4; ++kt) {
        float e = __expf(sb[kt][r] - mn);
        sb[kt][r] = e;
        sum += e;
      }
#pragma unroll
      for (int off = 1; off < 16; off <<= 1) sum += __shfl_xor(sum, off, 32);
      lrow[r] = lrow[r] * sc + sum;
#pragma unroll
      for (int nt = 0; nt < 4; ++nt) ov[nt][r] *= sc;
    }
#pragma unroll
    for (int kt = 0; kt < 4; ++kt) {
      v8f pv = sb[kt];
#pragma unroll
      for (int r = 0; r < 8; ++r)
        pbuf[(mh + r) * 64 + kt * 16 + nn] = (_Float16)pv[r];
    }
#pragma unroll
    for (int c = 0; c < 2; ++c) {
      v16h a = load_a_frag(pbuf + c * 32, 64);
#pragma unroll
      for (int nt = 0; nt < 4; ++nt) {
        v16h bb = load_bt_frag(vsbT + (nt * 16) * LDV_G + c * 32, LDV_G);
        ov[nt] = wmma16(a, bb, ov[nt]);
      }
    }
    __syncthreads();
  }
#pragma unroll
  for (int nt = 0; nt < 4; ++nt)
#pragma unroll
    for (int r = 0; r < 8; ++r)
      obuf[(mh + r) * 64 + nt * 16 + nn] = ov[nt][r] / lrow[r];
  for (int idx = lane; idx < 256; idx += 32) {
    int row = idx >> 4, g = idx & 15;
    int s = qb0 + w * 16 + row;
    const float* M = Gi + (size_t)(b * 8 + (s >> 8)) * 16;
    float o0 = obuf[row * 64 + g * 4 + 0];
    float o1 = obuf[row * 64 + g * 4 + 1];
    float o2 = obuf[row * 64 + g * 4 + 2];
    float o3 = obuf[row * 64 + g * 4 + 3];
#pragma unroll
    for (int i = 0; i < 4; ++i)
      y[(size_t)(b * 2048 + s) * 768 + h * 64 + g * 4 + i] =
          (_Float16)(M[i * 4 + 0] * o0 + M[i * 4 + 1] * o1 + M[i * 4 + 2] * o2 + M[i * 4 + 3] * o3);
  }
}

// ---------------------------------------------------------------------------
// Host launcher
// ---------------------------------------------------------------------------
extern "C" void kernel_launch(void* const* d_in, const int* in_sizes, int n_in,
                              void* d_out, int out_size, void* d_ws, size_t ws_size,
                              hipStream_t stream) {
  (void)in_sizes; (void)n_in; (void)out_size; (void)ws_size;
  const float* x        = (const float*)d_in[0];
  const float* viewmats = (const float*)d_in[1];
  const float* Ks       = (const float*)d_in[2];
  const float* ln1_g = (const float*)d_in[3];
  const float* ln1_b = (const float*)d_in[4];
  const float* Wqkv_l = (const float*)d_in[5];
  const float* Wproj_l = (const float*)d_in[6];
  const float* ln2_g = (const float*)d_in[7];
  const float* ln2_b = (const float*)d_in[8];
  const float* W1_l = (const float*)d_in[9];
  const float* b1_l = (const float*)d_in[10];
  const float* W2_l = (const float*)d_in[11];
  const float* b2_l = (const float*)d_in[12];
  const float* ln3_g = (const float*)d_in[13];
  const float* ln3_b = (const float*)d_in[14];
  const float* Wqkv_g = (const float*)d_in[15];
  const float* Wproj_g = (const float*)d_in[16];
  const float* ln4_g = (const float*)d_in[17];
  const float* ln4_b = (const float*)d_in[18];
  const float* W1_g = (const float*)d_in[19];
  const float* b1_g = (const float*)d_in[20];
  const float* W2_g = (const float*)d_in[21];
  const float* b2_g = (const float*)d_in[22];

  const int M = 4096, D = 768;
  size_t off = 0;
  char* base = (char*)d_ws;
  auto alloc = [&](size_t bytes) -> void* {
    void* p = base + off;
    off += (bytes + 255) & ~(size_t)255;
    return p;
  };
  _Float16* wqkv_l16  = (_Float16*)alloc((size_t)768 * 2304 * 2);
  _Float16* wproj_l16 = (_Float16*)alloc((size_t)768 * 768 * 2);
  _Float16* w1_l16    = (_Float16*)alloc((size_t)768 * 1536 * 2);
  _Float16* w2_l16    = (_Float16*)alloc((size_t)1536 * 768 * 2);
  _Float16* wqkv_g16  = (_Float16*)alloc((size_t)768 * 2304 * 2);
  _Float16* wproj_g16 = (_Float16*)alloc((size_t)768 * 768 * 2);
  _Float16* w1_g16    = (_Float16*)alloc((size_t)768 * 3072 * 2);
  _Float16* w2_g16    = (_Float16*)alloc((size_t)3072 * 768 * 2);
  float*    xbuf = (float*)alloc((size_t)M * D * 4);
  _Float16* xn16 = (_Float16*)alloc((size_t)M * D * 2);
  float*    qkv  = (float*)alloc((size_t)M * 2304 * 4);
  _Float16* y16  = (_Float16*)alloc((size_t)M * D * 2);
  _Float16* h16  = (_Float16*)alloc((size_t)M * 3072 * 2);
  float*    Gmat = (float*)alloc(16 * 16 * 4);
  float*    Gima = (float*)alloc(16 * 16 * 4);

  auto cvt_t = [&](const float* s, _Float16* d, int K, int N) {
    k_cvt_f16_t<<<(K * N + 255) / 256, 256, 0, stream>>>(s, d, K, N);
  };
  cvt_t(Wqkv_l, wqkv_l16, 768, 2304);
  cvt_t(Wproj_l, wproj_l16, 768, 768);
  cvt_t(W1_l, w1_l16, 768, 1536);
  cvt_t(W2_l, w2_l16, 1536, 768);
  cvt_t(Wqkv_g, wqkv_g16, 768, 2304);
  cvt_t(Wproj_g, wproj_g16, 768, 768);
  cvt_t(W1_g, w1_g16, 768, 3072);
  cvt_t(W2_g, w2_g16, 3072, 768);

  (void)hipMemcpyAsync(xbuf, x, (size_t)M * D * 4, hipMemcpyDeviceToDevice, stream);
  k_gmats<<<1, 32, 0, stream>>>(viewmats, Ks, Gmat, Gima);

  const size_t SMEM_LOCAL =
      (size_t)(2 * 256 * 64 + 64 * LDV_L + 8 * 16 * 32) * 2;            // ~105 KB
  const size_t SMEM_GLOBAL =
      (size_t)(128 * 64 + 64 * 64 + 64 * LDV_G + 8 * 16 * 64) * 2
      + (size_t)(8 * 16 * 64) * 4;                                      // ~81 KB

  // ---- local per-view block ----
  k_ln<<<M / 8, 256, 0, stream>>>(xbuf, ln1_g, ln1_b, xn16);
  k_gemm<<<dim3(2304 / 128, M / 64), 256, 0, stream>>>(xn16, wqkv_l16, nullptr, nullptr,
                                                       qkv, M, 2304, 768, 0);
  k_attn_local<<<192, 256, SMEM_LOCAL, stream>>>(qkv, y16);
  k_gemm<<<dim3(768 / 128, M / 64), 256, 0, stream>>>(y16, wproj_l16, nullptr, xbuf,
                                                      xbuf, M, 768, 768, 4);
  k_ln<<<M / 8, 256, 0, stream>>>(xbuf, ln2_g, ln2_b, xn16);
  k_gemm<<<dim3(1536 / 128, M / 64), 256, 0, stream>>>(xn16, w1_l16, b1_l, nullptr,
                                                       h16, M, 1536, 768, 8 | 1 | 2);
  k_gemm<<<dim3(768 / 128, M / 64), 256, 0, stream>>>(h16, w2_l16, b2_l, xbuf,
                                                      xbuf, M, 768, 1536, 8 | 4);
  // ---- global PRoPE block ----
  k_ln<<<M / 8, 256, 0, stream>>>(xbuf, ln3_g, ln3_b, xn16);
  k_gemm<<<dim3(2304 / 128, M / 64), 256, 0, stream>>>(xn16, wqkv_g16, nullptr, nullptr,
                                                       qkv, M, 2304, 768, 0);
  k_attn_global<<<384, 256, SMEM_GLOBAL, stream>>>(qkv, Gmat, Gima, y16);
  k_gemm<<<dim3(768 / 128, M / 64), 256, 0, stream>>>(y16, wproj_g16, nullptr, xbuf,
                                                      xbuf, M, 768, 768, 4);
  k_ln<<<M / 8, 256, 0, stream>>>(xbuf, ln4_g, ln4_b, xn16);
  k_gemm<<<dim3(3072 / 128, M / 64), 256, 0, stream>>>(xn16, w1_g16, b1_g, nullptr,
                                                       h16, M, 3072, 768, 8 | 1 | 2);
  k_gemm<<<dim3(768 / 128, M / 64), 256, 0, stream>>>(h16, w2_g16, b2_g, xbuf,
                                                      (float*)d_out, M, 768, 3072, 8 | 4);
}
